// RnnNet_13322988552223
// MI455X (gfx1250) — matrix operand order repaired and checked by reading, not compiled, not run on hardware
//
#include <hip/hip_runtime.h>
#include <hip/hip_bf16.h>

typedef _Float16 half8_t  __attribute__((ext_vector_type(8)));
typedef _Float16 half16_t __attribute__((ext_vector_type(16)));
typedef float    v8f_t    __attribute__((ext_vector_type(8)));
typedef float    f4_t     __attribute__((ext_vector_type(4)));

#define NWAVES 4
#define NT     (NWAVES * 32)
#define P_IH0  40   // pitch (halves) for W_ih0: K=28 padded to 32, row pitch 40
#define P_HH   72   // pitch (halves) for 64-wide weights / h staging

__device__ __forceinline__ v8f_t wmma_f16(half16_t a, half16_t b, v8f_t c) {
    return __builtin_amdgcn_wmma_f32_16x16x32_f16(false, a, false, b, (short)0, c,
                                                  false, false);
}

// Scheduling fence: keep the unrolled tg iterations from being merged by the
// scheduler (bounds VGPR pressure without giving up static register indexing).
__device__ __forceinline__ void sched_fence() {
#if __has_builtin(__builtin_amdgcn_sched_barrier)
    __builtin_amdgcn_sched_barrier(0);
#endif
}

// A-fragment (16x32 f16, M x K): lane L holds M=L%16;
// halves 0..7  = K (kfrag + 8*(L/16) + 0..7)
// halves 8..15 = K (kfrag + 8*(L/16) + 16 + 0..7)
__device__ __forceinline__ half16_t ldsA(const _Float16* base, int mrow, int k0sub, int kfrag) {
    const _Float16* p = base + mrow * P_HH + kfrag + k0sub;
    half8_t lo = *(const half8_t*)(p);
    half8_t hi = *(const half8_t*)(p + 16);
    return __builtin_shufflevector(lo, hi, 0,1,2,3,4,5,6,7,8,9,10,11,12,13,14,15);
}

// B-fragment (32x16 f16, K x N): lane L holds N=L%16, halves 0..7 = K(kfrag + 8*(L/16) + 0..7), halves 8..15 = K(kfrag + 8*(L/16) + 16 + 0..7)
// B[k][n] = W[gateRow = g0+n][k]  (since gates = act @ W^T)
__device__ __forceinline__ half16_t ldsB(const _Float16* base, int pitch, int g0,
                                         int nlane, int ksel, int kfrag) {
    const _Float16* p = base + (g0 + nlane) * pitch + kfrag + (ksel >> 1);
    half8_t lo = *(const half8_t*)(p);
    half8_t hi = *(const half8_t*)(p + 16);
    return __builtin_shufflevector(lo, hi, 0,1,2,3,4,5,6,7,8,9,10,11,12,13,14,15);
}

// CDNA5 has V_TANH_F32 (TRANS pipe -> co-executes with XDL WMMA).
__device__ __forceinline__ float fast_tanh(float x) {
#if __has_builtin(__builtin_amdgcn_tanhf)
    return __builtin_amdgcn_tanhf(x);
#else
    return 1.0f - 2.0f * __builtin_amdgcn_rcpf(1.0f + __expf(2.0f * x));
#endif
}
__device__ __forceinline__ float fast_sigmoid(float x) {
#if __has_builtin(__builtin_amdgcn_tanhf)
    return fmaf(0.5f, __builtin_amdgcn_tanhf(0.5f * x), 0.5f);
#else
    return __builtin_amdgcn_rcpf(1.0f + __expf(-x));
#endif
}

__global__ void __launch_bounds__(NT)
lstm2_wmma_kernel(const float* __restrict__ x,
                  const float* __restrict__ Wih0, const float* __restrict__ Whh0,
                  const float* __restrict__ bih0, const float* __restrict__ bhh0,
                  const float* __restrict__ Wih1, const float* __restrict__ Whh1,
                  const float* __restrict__ bih1, const float* __restrict__ bhh1,
                  const float* __restrict__ Wlin, const float* __restrict__ blin,
                  float* __restrict__ out, int B) {
    // ---- LDS: f16 weights (padded pitch for conflict-free b128 fragment loads) ----
    __shared__ __align__(16) _Float16 sWih0[256 * P_IH0];   // 20.0 KB
    __shared__ __align__(16) _Float16 sWhh0[256 * P_HH];    // 36.0 KB
    __shared__ __align__(16) _Float16 sWih1[256 * P_HH];    // 36.0 KB
    __shared__ __align__(16) _Float16 sWhh1[256 * P_HH];    // 36.0 KB
    __shared__ float sB0[256], sB1[256];
    __shared__ float sWlin[10 * 64];
    __shared__ float sblin[16];
    __shared__ __align__(16) _Float16 sH1[NWAVES][16 * P_HH]; // per-wave h1 staging
    __shared__ __align__(16) _Float16 sH2[NWAVES][16 * P_HH]; // per-wave h2 staging

    const int tid = threadIdx.x;

    // ---- cooperative weight load + f32->f16 convert ----
    for (int i = tid; i < 256 * 32; i += NT) {
        int r = i >> 5, k = i & 31;
        sWih0[r * P_IH0 + k] = (_Float16)((k < 28) ? Wih0[r * 28 + k] : 0.0f);
    }
    for (int i = tid; i < 256 * 64; i += NT) {
        int r = i >> 6, k = i & 63;
        sWhh0[r * P_HH + k] = (_Float16)Whh0[i];
        sWih1[r * P_HH + k] = (_Float16)Wih1[i];
        sWhh1[r * P_HH + k] = (_Float16)Whh1[i];
    }
    for (int i = tid; i < 256; i += NT) {
        sB0[i] = bih0[i] + bhh0[i];
        sB1[i] = bih1[i] + bhh1[i];
    }
    for (int i = tid; i < 640; i += NT) sWlin[i] = Wlin[i];
    for (int i = tid; i < 10;  i += NT) sblin[i] = blin[i];
    __syncthreads();

    const int wave   = tid >> 5;
    const int lane   = tid & 31;
    const int nlane  = lane & 15;          // N / M-row index (0..15)
    const int khalf  = lane >> 4;          // 0 or 1
    const int k0sub  = khalf * 8;          // A-fragment per-lane K sub-offset
    const int ksel   = khalf * 16;         // B-fragment per-lane K sub-offset
    const int khalf8 = khalf * 8;          // D-layout M offset

    const int sampleBase = (blockIdx.x * NWAVES + wave) * 16;
    if (sampleBase >= B) return;           // no syncs past this point

    _Float16* sH1w = sH1[wave];
    _Float16* sH2w = sH2[wave];
    for (int i = lane; i < 16 * P_HH; i += 32) { sH1w[i] = (_Float16)0.0f; sH2w[i] = (_Float16)0.0f; }

    v8f_t c0s[4], c1s[4];
    #pragma unroll
    for (int tg = 0; tg < 4; ++tg) { c0s[tg] = (v8f_t){}; c1s[tg] = (v8f_t){}; }

    // this lane's x row (A-layout: lane covers sample M = lane%16); rows are 16B aligned
    const float* xrow = x + (size_t)(sampleBase + nlane) * 784;

    #pragma unroll 1
    for (int t = 0; t < 28; ++t) {
        const float* xt = xrow + t * 28;          // 16B aligned (112*t bytes)
        if (t < 27) __builtin_prefetch(xt + 28, 0, 3);   // global_prefetch next step

        // x A-fragment, K=28 zero-padded to 32 (vector b128 loads; tail masked)
        const f4_t* xv = (const f4_t*)xt;
        f4_t q0 = xv[khalf * 2];                  // k0sub + 0..3
        f4_t q1 = xv[khalf * 2 + 1];              // k0sub + 4..7
        f4_t q2 = xv[khalf * 2 + 4];              // k0sub + 16..19
        f4_t q3 = khalf ? (f4_t){0.f, 0.f, 0.f, 0.f} : xv[5];  // 20..23 or pad 28..31
        half16_t ax;
        #pragma unroll
        for (int j = 0; j < 4; ++j) {
            ax[j]      = (_Float16)q0[j];
            ax[4 + j]  = (_Float16)q1[j];
            ax[8 + j]  = (_Float16)q2[j];
            ax[12 + j] = (_Float16)q3[j];
        }

        // previous h1 A-fragments (K=0..31, 32..63)
        half16_t a10 = ldsA(sH1w, nlane, k0sub, 0);
        half16_t a11 = ldsA(sH1w, nlane, k0sub, 32);

        // ================= layer 0 =================
        #pragma unroll
        for (int tg = 0; tg < 4; ++tg) {     // hidden tile; gate tiles tg,tg+4,tg+8,tg+12
            float bi = sB0[(tg     ) * 16 + nlane];
            float bf = sB0[(tg +  4) * 16 + nlane];
            float bg = sB0[(tg +  8) * 16 + nlane];
            float bo = sB0[(tg + 12) * 16 + nlane];
            v8f_t acc[4];
            #pragma unroll
            for (int q = 0; q < 4; ++q) {
                int g0 = (tg + q * 4) * 16;
                v8f_t a = (v8f_t){};     // inline-0 C operand
                a = wmma_f16(ax,  ldsB(sWih0, P_IH0, g0, nlane, ksel, 0),  a);
                a = wmma_f16(a10, ldsB(sWhh0, P_HH,  g0, nlane, ksel, 0),  a);
                a = wmma_f16(a11, ldsB(sWhh0, P_HH,  g0, nlane, ksel, 32), a);
                acc[q] = a;
            }
            #pragma unroll
            for (int r = 0; r < 8; ++r) {
                float iv = fast_sigmoid(acc[0][r] + bi);
                float fv = fast_sigmoid(acc[1][r] + bf);
                float gv = fast_tanh   (acc[2][r] + bg);
                float ov = fast_sigmoid(acc[3][r] + bo);
                float cv = fv * c0s[tg][r] + iv * gv;
                c0s[tg][r] = cv;
                float hv = ov * fast_tanh(cv);
                // D-layout -> [M][K] staging: M = r + khalf*8, K = tg*16 + nlane
                sH1w[(r + khalf8) * P_HH + tg * 16 + nlane] = (_Float16)hv;
            }
            sched_fence();               // bound scheduler hoisting across tg iters
        }

        // new h1 and previous h2 A-fragments
        half16_t b10 = ldsA(sH1w, nlane, k0sub, 0);
        half16_t b11 = ldsA(sH1w, nlane, k0sub, 32);
        half16_t a20 = ldsA(sH2w, nlane, k0sub, 0);
        half16_t a21 = ldsA(sH2w, nlane, k0sub, 32);

        // ================= layer 1 =================
        #pragma unroll
        for (int tg = 0; tg < 4; ++tg) {
            float bi = sB1[(tg     ) * 16 + nlane];
            float bf = sB1[(tg +  4) * 16 + nlane];
            float bg = sB1[(tg +  8) * 16 + nlane];
            float bo = sB1[(tg + 12) * 16 + nlane];
            v8f_t acc[4];
            #pragma unroll
            for (int q = 0; q < 4; ++q) {
                int g0 = (tg + q * 4) * 16;
                v8f_t a = (v8f_t){};
                a = wmma_f16(b10, ldsB(sWih1, P_HH, g0, nlane, ksel, 0),  a);
                a = wmma_f16(b11, ldsB(sWih1, P_HH, g0, nlane, ksel, 32), a);
                a = wmma_f16(a20, ldsB(sWhh1, P_HH, g0, nlane, ksel, 0),  a);
                a = wmma_f16(a21, ldsB(sWhh1, P_HH, g0, nlane, ksel, 32), a);
                acc[q] = a;
            }
            #pragma unroll
            for (int r = 0; r < 8; ++r) {
                float iv = fast_sigmoid(acc[0][r] + bi);
                float fv = fast_sigmoid(acc[1][r] + bf);
                float gv = fast_tanh   (acc[2][r] + bg);
                float ov = fast_sigmoid(acc[3][r] + bo);
                float cv = fv * c1s[tg][r] + iv * gv;
                c1s[tg][r] = cv;
                float hv = ov * fast_tanh(cv);
                sH2w[(r + khalf8) * P_HH + tg * 16 + nlane] = (_Float16)hv;
            }
            sched_fence();
        }
    }

    // ---- final linear: out[s][o] = h2_last[s] . Wlin[o] + blin[o] ----
    for (int idx = lane; idx < 16 * 10; idx += 32) {
        int s = idx / 10, o = idx % 10;
        const _Float16* hp = sH2w + s * P_HH;
        const float*    wp = sWlin + o * 64;
        float acc = sblin[o];
        #pragma unroll 8
        for (int k = 0; k < 64; ++k) acc += (float)hp[k] * wp[k];
        out[(size_t)(sampleBase + s) * 10 + o] = acc;
    }
}

extern "C" void kernel_launch(void* const* d_in, const int* in_sizes, int n_in,
                              void* d_out, int out_size, void* d_ws, size_t ws_size,
                              hipStream_t stream) {
    const float* x    = (const float*)d_in[0];
    const float* Wih0 = (const float*)d_in[1];
    const float* Whh0 = (const float*)d_in[2];
    const float* bih0 = (const float*)d_in[3];
    const float* bhh0 = (const float*)d_in[4];
    const float* Wih1 = (const float*)d_in[5];
    const float* Whh1 = (const float*)d_in[6];
    const float* bih1 = (const float*)d_in[7];
    const float* bhh1 = (const float*)d_in[8];
    const float* Wlin = (const float*)d_in[9];
    const float* blin = (const float*)d_in[10];

    int B = in_sizes[0] / 784;                 // 16384
    int blocks = (B + NWAVES * 16 - 1) / (NWAVES * 16);

    lstm2_wmma_kernel<<<blocks, NT, 0, stream>>>(
        x, Wih0, Whh0, bih0, bhh0, Wih1, Whh1, bih1, bhh1, Wlin, blin,
        (float*)d_out, B);
}
